// SelfAttention_82360292868817
// MI455X (gfx1250) — matrix operand order repaired and checked by reading, not compiled, not run on hardware
//
#include <hip/hip_runtime.h>
#include <hip/hip_bf16.h>

// ---------------------------------------------------------------------------
// Types for CDNA5 WMMA
// ---------------------------------------------------------------------------
typedef __bf16 bf16_t;
typedef bf16_t bf16x16 __attribute__((ext_vector_type(16)));
typedef float  floatx8 __attribute__((ext_vector_type(8)));

__device__ __forceinline__ floatx8 wmma_bf16(bf16x16 a, bf16x16 b, floatx8 c) {
    // D = A(16x32 bf16) * B(32x16 bf16) + C(16x16 f32)
    return __builtin_amdgcn_wmma_f32_16x16x32_bf16(
        /*neg_a=*/false, a, /*neg_b=*/false, b,
        /*c_mod=*/(short)0, c, /*reuse_a=*/false, /*reuse_b=*/false);
}

// CDNA5 async copy: global -> LDS, tracked by ASYNCcnt, no VGPR data movement.
// INST_OFFSET applies to both the LDS and global address (ISA 08 §4.4).
#define ASYNC_CP16(lds32, g64, OFFSTR)                                        \
    asm volatile("global_load_async_to_lds_b128 %0, %1, off offset:" OFFSTR   \
                 :: "v"(lds32), "v"(g64) : "memory")
#define WAIT_ASYNC(NSTR) \
    asm volatile("s_wait_asynccnt " NSTR ::: "memory")

// Problem constants
static constexpr int BATCH = 2;
static constexpr int SEQ   = 2048;
static constexpr int DMODEL= 2048;
static constexpr int NH    = 16;
static constexpr int HD    = 128;
static constexpr int HALF  = 64;
static constexpr float EPSF = 1.1920928955078125e-07f;

// ---------------------------------------------------------------------------
// Kernel 1: fp32 -> bf16 cast (vectorized x4)
// ---------------------------------------------------------------------------
__global__ __launch_bounds__(256) void cast_f32_bf16(
    const float* __restrict__ src, bf16_t* __restrict__ dst, int n)
{
    int i = (blockIdx.x * 256 + threadIdx.x) * 4;
    if (i + 3 < n) {
        float4 f = *(const float4*)(src + i);
        dst[i + 0] = (bf16_t)f.x;
        dst[i + 1] = (bf16_t)f.y;
        dst[i + 2] = (bf16_t)f.z;
        dst[i + 3] = (bf16_t)f.w;
    }
}

// ---------------------------------------------------------------------------
// Kernel 2: C[M,N](f32) = A[M,K](bf16 rm) * B[N,K](bf16 rm)^T
// Block 256 thr = 8 waves. Tile BM=256 x BN=128, BK=32; wave -> 32 rows.
// Double-buffered LDS staged entirely with global_load_async_to_lds_b128.
// Per k-step per wave: 2 A-frags + 8 B-frags -> 16 v_wmma back-to-back.
// ---------------------------------------------------------------------------
#define BM 256
#define BN 128
#define BKK 32

__global__ __launch_bounds__(256) void gemm_nt_bf16(
    const bf16_t* __restrict__ A, const bf16_t* __restrict__ B,
    float* __restrict__ C, int M, int N, int K)
{
    __shared__ bf16_t As[2][BM * BKK];   // 2 x 16 KB
    __shared__ bf16_t Bs[2][BN * BKK];   // 2 x  8 KB

    const int tid  = threadIdx.x;
    const int lane = tid & 31;
    const int wave = tid >> 5;
    const int m0 = blockIdx.y * BM;
    const int n0 = blockIdx.x * BN;

    // Staging: thread t moves A row (m0+t), 64B; and 32B of B row (n0 + t/2).
    const bf16_t* gA = A + (size_t)(m0 + tid) * K;
    const bf16_t* gB = B + (size_t)(n0 + (tid >> 1)) * K + (tid & 1) * 16;
    const unsigned laA[2] = { (unsigned)(size_t)&As[0][tid * BKK],
                              (unsigned)(size_t)&As[1][tid * BKK] };
    const unsigned laB[2] = { (unsigned)(size_t)&Bs[0][(tid >> 1) * BKK + (tid & 1) * 16],
                              (unsigned)(size_t)&Bs[1][(tid >> 1) * BKK + (tid & 1) * 16] };

    auto stage = [&](int buf, int k0) {
        unsigned long long ga = (unsigned long long)(size_t)(gA + k0);
        unsigned long long gb = (unsigned long long)(size_t)(gB + k0);
        ASYNC_CP16(laA[buf], ga, "0");
        ASYNC_CP16(laA[buf], ga, "16");
        ASYNC_CP16(laA[buf], ga, "32");
        ASYNC_CP16(laA[buf], ga, "48");
        ASYNC_CP16(laB[buf], gb, "0");
        ASYNC_CP16(laB[buf], gb, "16");
    };

    floatx8 zero8 = {0.f,0.f,0.f,0.f,0.f,0.f,0.f,0.f};
    floatx8 acc[2][8];
#pragma unroll
    for (int mt = 0; mt < 2; ++mt)
#pragma unroll
        for (int nt = 0; nt < 8; ++nt) acc[mt][nt] = zero8;

    union u16x { bf16x16 v; uint4 q[2]; };

    stage(0, 0);
    for (int k0 = 0; k0 < K; k0 += BKK) {
        const int cur = (k0 / BKK) & 1;
        if (k0 + BKK < K) {
            stage(cur ^ 1, k0 + BKK);       // prefetch next tile into other buf
            WAIT_ASYNC("6");                // 6 newest = just issued; prev tile done
        } else {
            WAIT_ASYNC("0");
        }
        __syncthreads();

        // A fragments: rows 32*wave + (lane&15) and +16
        const int am = wave * 32 + (lane & 15);
        const int ak = (lane >> 4) * 8;
        u16x a0, a1;
        a0.q[0] = *(const uint4*)&As[cur][am * BKK + ak];
        a0.q[1] = *(const uint4*)&As[cur][am * BKK + ak + 16];
        a1.q[0] = *(const uint4*)&As[cur][(am + 16) * BKK + ak];
        a1.q[1] = *(const uint4*)&As[cur][(am + 16) * BKK + ak + 16];

        // Preload all 8 B fragments, then issue 16 WMMAs back-to-back.
        u16x bfr[8];
#pragma unroll
        for (int nt = 0; nt < 8; ++nt) {
            const int bn = nt * 16 + (lane & 15);
            const int bk = (lane >> 4) * 8;
            bfr[nt].q[0] = *(const uint4*)&Bs[cur][bn * BKK + bk];
            bfr[nt].q[1] = *(const uint4*)&Bs[cur][bn * BKK + bk + 16];
        }
#pragma unroll
        for (int nt = 0; nt < 8; ++nt)
            acc[0][nt] = wmma_bf16(a0.v, bfr[nt].v, acc[0][nt]);
#pragma unroll
        for (int nt = 0; nt < 8; ++nt)
            acc[1][nt] = wmma_bf16(a1.v, bfr[nt].v, acc[1][nt]);
        __syncthreads();
    }

    // C layout: vgpr j, lanes 0-15 -> M=j, N=lane; lanes 16-31 -> M=8+j
    const int ccol = n0 + (lane & 15);
#pragma unroll
    for (int mt = 0; mt < 2; ++mt) {
        const int crow = m0 + wave * 32 + mt * 16 + (lane >> 4) * 8;
#pragma unroll
        for (int nt = 0; nt < 8; ++nt)
#pragma unroll
            for (int j = 0; j < 8; ++j)
                C[(size_t)(crow + j) * N + ccol + nt * 16] = acc[mt][nt][j];
    }
}

// ---------------------------------------------------------------------------
// Kernel 3: RoPE + RMS-norm (*1.2) on Q,K; cast Q,K,V -> bf16.
// One wave per (row, head): 32 lanes x 4 elements = 128 dims.
// ---------------------------------------------------------------------------
__global__ __launch_bounds__(256) void rope_norm_cast(
    const float* __restrict__ Qf, const float* __restrict__ Kf,
    const float* __restrict__ Vf,
    const float* __restrict__ cosT, const float* __restrict__ sinT,
    bf16_t* __restrict__ Qb, bf16_t* __restrict__ Kb, bf16_t* __restrict__ Vb)
{
    const int wid  = blockIdx.x * 8 + (threadIdx.x >> 5); // (row*NH + h)
    const int lane = threadIdx.x & 31;
    const int h    = wid & (NH - 1);
    const int row  = wid >> 4;            // b*SEQ + s   (NH==16)
    const int s    = row & (SEQ - 1);
    const size_t base = (size_t)row * DMODEL + (size_t)h * HD;
    const int i0 = lane * 2, i1 = lane * 2 + 1;

    const float ca = cosT[s * HALF + i0], cb = cosT[s * HALF + i1];
    const float sa = sinT[s * HALF + i0], sb = sinT[s * HALF + i1];

    // ---- Q ----
    {
        float x1a = Qf[base + i0],        x1b = Qf[base + i1];
        float x2a = Qf[base + i0 + HALF], x2b = Qf[base + i1 + HALF];
        float ra = x1a * ca + x2a * sa;
        float rb = x1b * cb + x2b * sb;
        float rc = -x1a * sa + x2a * ca;
        float rd = -x1b * sb + x2b * cb;
        float ss = ra*ra + rb*rb + rc*rc + rd*rd;
#pragma unroll
        for (int m = 16; m; m >>= 1) ss += __shfl_xor(ss, m, 32);
        float r = rsqrtf(ss * (1.0f / HD) + EPSF) * 1.2f;
        Qb[base + i0]        = (bf16_t)(ra * r);
        Qb[base + i1]        = (bf16_t)(rb * r);
        Qb[base + i0 + HALF] = (bf16_t)(rc * r);
        Qb[base + i1 + HALF] = (bf16_t)(rd * r);
    }
    // ---- K ----
    {
        float x1a = Kf[base + i0],        x1b = Kf[base + i1];
        float x2a = Kf[base + i0 + HALF], x2b = Kf[base + i1 + HALF];
        float ra = x1a * ca + x2a * sa;
        float rb = x1b * cb + x2b * sb;
        float rc = -x1a * sa + x2a * ca;
        float rd = -x1b * sb + x2b * cb;
        float ss = ra*ra + rb*rb + rc*rc + rd*rd;
#pragma unroll
        for (int m = 16; m; m >>= 1) ss += __shfl_xor(ss, m, 32);
        float r = rsqrtf(ss * (1.0f / HD) + EPSF) * 1.2f;
        Kb[base + i0]        = (bf16_t)(ra * r);
        Kb[base + i1]        = (bf16_t)(rb * r);
        Kb[base + i0 + HALF] = (bf16_t)(rc * r);
        Kb[base + i1 + HALF] = (bf16_t)(rd * r);
    }
    // ---- V passthrough ----
    Vb[base + i0]        = (bf16_t)Vf[base + i0];
    Vb[base + i1]        = (bf16_t)Vf[base + i1];
    Vb[base + i0 + HALF] = (bf16_t)Vf[base + i0 + HALF];
    Vb[base + i1 + HALF] = (bf16_t)Vf[base + i1 + HALF];
}

// ---------------------------------------------------------------------------
// Kernel 4: flash attention per (b, h, 128 query rows).
// 8 waves; wave w owns q rows [16w,16w+16). Key blocks of 32.
// K staged via async-to-LDS; V transposed through VGPRs; next block prefetched.
// ---------------------------------------------------------------------------
__global__ __launch_bounds__(256) void flash_attn(
    const bf16_t* __restrict__ Qb, const bf16_t* __restrict__ Kb,
    const bf16_t* __restrict__ Vb, bf16_t* __restrict__ Yb)
{
    __shared__ bf16_t Ks[32 * 128];     // [key][d]
    __shared__ bf16_t Vt[128 * 32];     // [d][key] (transposed)
    __shared__ bf16_t Ps[8][16 * 32];   // per-wave P tile (16 rows x 32 keys)

    const int tid  = threadIdx.x;
    const int lane = tid & 31;
    const int wave = tid >> 5;
    const int b  = blockIdx.z;
    const int h  = blockIdx.y;
    const int q0 = blockIdx.x * 128;
    const size_t batchoff = (size_t)b * SEQ * DMODEL;
    const size_t headoff  = (size_t)h * HD;

    union u16x { bf16x16 v; uint4 q[2]; };

    // Q fragments: 16 rows x 128 dims = 4 K-chunks of 32
    const int    qm    = q0 + wave * 16 + (lane & 15);
    const size_t qbase = batchoff + (size_t)qm * DMODEL + headoff + (size_t)((lane >> 4) * 8);
    u16x qf[4];
#pragma unroll
    for (int c = 0; c < 4; ++c) {
        qf[c].q[0] = *(const uint4*)(Qb + qbase + c * 32);
        qf[c].q[1] = *(const uint4*)(Qb + qbase + c * 32 + 16);
    }

    floatx8 zero8 = {0.f,0.f,0.f,0.f,0.f,0.f,0.f,0.f};
    floatx8 o[8];
#pragma unroll
    for (int i = 0; i < 8; ++i) o[i] = zero8;
    float rmax[8], rsum[8];
#pragma unroll
    for (int j = 0; j < 8; ++j) { rmax[j] = -1.0e30f; rsum[j] = 0.0f; }

    // staging geometry: thread -> (key row r, 16-elem col chunk cth)
    const int r   = tid >> 3;
    const int cth = (tid & 7) * 16;
    const unsigned lk = (unsigned)(size_t)&Ks[r * 128 + cth];
    const bf16_t* gK = Kb + batchoff + headoff + (size_t)r * DMODEL + cth;
    const bf16_t* gV = Vb + batchoff + headoff + (size_t)r * DMODEL + cth;

    // fold 1/sqrt(Hd) and log2(e) so softmax runs in base 2
    const float scale = 0.08838834764831845f * 1.4426950408889634f;

    for (int kb = 0; kb < SEQ; kb += 32) {
        // ---- stage K block [32][128] via async copy ----
        {
            unsigned long long gk = (unsigned long long)(size_t)(gK + (size_t)kb * DMODEL);
            ASYNC_CP16(lk, gk, "0");
            ASYNC_CP16(lk, gk, "16");
        }
        // ---- stage V block transposed -> Vt[128][32] ----
        {
            const uint4* g = (const uint4*)(gV + (size_t)kb * DMODEL);
            union { uint4 q[2]; bf16_t e[16]; } vv;
            vv.q[0] = g[0]; vv.q[1] = g[1];
#pragma unroll
            for (int e = 0; e < 16; ++e)
                Vt[(cth + e) * 32 + r] = vv.e[e];
        }
        // prefetch next key block while this one computes
        if (kb + 32 < SEQ) {
            __builtin_prefetch(gK + (size_t)(kb + 32) * DMODEL, 0, 0);
            __builtin_prefetch(gV + (size_t)(kb + 32) * DMODEL, 0, 0);
        }
        WAIT_ASYNC("0");
        __syncthreads();

        // ---- scores: two 16x16 tiles over the 32-key block ----
        floatx8 sacc[2] = {zero8, zero8};
#pragma unroll
        for (int t = 0; t < 2; ++t) {
#pragma unroll
            for (int c = 0; c < 4; ++c) {
                const int kn = t * 16 + (lane & 15);
                const int kk = c * 32 + (lane >> 4) * 8;
                u16x kf;
                kf.q[0] = *(const uint4*)&Ks[kn * 128 + kk];
                kf.q[1] = *(const uint4*)&Ks[kn * 128 + kk + 16];
                sacc[t] = wmma_bf16(qf[c].v, kf.v, sacc[t]);
            }
        }

        // ---- online softmax (row stats live in the 16-lane half-groups) ----
        float pscale[8];
#pragma unroll
        for (int j = 0; j < 8; ++j) {
            float s0 = sacc[0][j] * scale;
            float s1 = sacc[1][j] * scale;
            float mx = fmaxf(s0, s1);
#pragma unroll
            for (int m = 8; m; m >>= 1) mx = fmaxf(mx, __shfl_xor(mx, m, 32));
            float nm   = fmaxf(rmax[j], mx);
            float corr = __builtin_exp2f(rmax[j] - nm);
            float p0   = __builtin_exp2f(s0 - nm);
            float p1   = __builtin_exp2f(s1 - nm);
            float ps   = p0 + p1;
#pragma unroll
            for (int m = 8; m; m >>= 1) ps += __shfl_xor(ps, m, 32);
            rsum[j] = rsum[j] * corr + ps;
            rmax[j] = nm;
            pscale[j] = corr;
            // spill P tile to per-wave LDS (C-layout -> row-major 16x32)
            const int prow = (lane >> 4) * 8 + j;
            const int pcol = lane & 15;
            Ps[wave][prow * 32 + pcol]      = (bf16_t)p0;
            Ps[wave][prow * 32 + pcol + 16] = (bf16_t)p1;
        }
#pragma unroll
        for (int n = 0; n < 8; ++n)
#pragma unroll
            for (int j = 0; j < 8; ++j)
                o[n][j] *= pscale[j];

        // wave-local LDS RAW: wait for our ds stores before A-frag reload
        asm volatile("s_wait_dscnt 0" ::: "memory");
        __builtin_amdgcn_wave_barrier();

        // ---- reload P as A-fragment (16x32) ----
        u16x pf;
        {
            const int pm = lane & 15;
            const int pk = (lane >> 4) * 8;
            pf.q[0] = *(const uint4*)&Ps[wave][pm * 32 + pk];
            pf.q[1] = *(const uint4*)&Ps[wave][pm * 32 + pk + 16];
        }

        // ---- o += P @ V over 8 d-tiles ----
#pragma unroll
        for (int n = 0; n < 8; ++n) {
            const int vd = n * 16 + (lane & 15);
            const int vk = (lane >> 4) * 8;
            u16x vf;
            vf.q[0] = *(const uint4*)&Vt[vd * 32 + vk];
            vf.q[1] = *(const uint4*)&Vt[vd * 32 + vk + 16];
            o[n] = wmma_bf16(pf.v, vf.v, o[n]);
        }
        __syncthreads();
    }

    // ---- normalize and emit bf16 y ----
#pragma unroll
    for (int j = 0; j < 8; ++j) rsum[j] = 1.0f / rsum[j];
    const int om = q0 + wave * 16 + (lane >> 4) * 8;
#pragma unroll
    for (int n = 0; n < 8; ++n)
#pragma unroll
        for (int j = 0; j < 8; ++j)
            Yb[batchoff + (size_t)(om + j) * DMODEL + headoff + n * 16 + (lane & 15)]
                = (bf16_t)(o[n][j] * rsum[j]);
}

// ---------------------------------------------------------------------------
// Host-side launcher
// ---------------------------------------------------------------------------
extern "C" void kernel_launch(void* const* d_in, const int* in_sizes, int n_in,
                              void* d_out, int out_size, void* d_ws, size_t ws_size,
                              hipStream_t stream)
{
    const float* x    = (const float*)d_in[0];
    const float* cosT = (const float*)d_in[1];
    const float* sinT = (const float*)d_in[2];
    const float* Wq   = (const float*)d_in[3];
    const float* Wk   = (const float*)d_in[4];
    const float* Wv   = (const float*)d_in[5];
    const float* Wo   = (const float*)d_in[6];
    float* out = (float*)d_out;

    const size_t NX = (size_t)BATCH * SEQ * DMODEL;   // 8M elems
    const size_t NW = (size_t)DMODEL * DMODEL;        // 4M elems

    char* p = (char*)d_ws;
    bf16_t* xb  = (bf16_t*)p; p += NX * sizeof(bf16_t);
    bf16_t* wqb = (bf16_t*)p; p += NW * sizeof(bf16_t);
    bf16_t* wkb = (bf16_t*)p; p += NW * sizeof(bf16_t);
    bf16_t* wvb = (bf16_t*)p; p += NW * sizeof(bf16_t);
    bf16_t* wob = (bf16_t*)p; p += NW * sizeof(bf16_t);
    float*  Qf  = (float*)p;  p += NX * sizeof(float);
    float*  Kf  = (float*)p;  p += NX * sizeof(float);
    float*  Vf  = (float*)p;  p += NX * sizeof(float);
    bf16_t* Qb  = (bf16_t*)p; p += NX * sizeof(bf16_t);
    bf16_t* Kb  = (bf16_t*)p; p += NX * sizeof(bf16_t);
    bf16_t* Vb  = (bf16_t*)p; p += NX * sizeof(bf16_t);
    bf16_t* Yb  = (bf16_t*)Qf;   // Qf dead after rope_norm_cast; reuse

    // 1) fp32 -> bf16 casts
    cast_f32_bf16<<<(int)(NX / 1024), 256, 0, stream>>>(x,  xb,  (int)NX);
    cast_f32_bf16<<<(int)(NW / 1024), 256, 0, stream>>>(Wq, wqb, (int)NW);
    cast_f32_bf16<<<(int)(NW / 1024), 256, 0, stream>>>(Wk, wkb, (int)NW);
    cast_f32_bf16<<<(int)(NW / 1024), 256, 0, stream>>>(Wv, wvb, (int)NW);
    cast_f32_bf16<<<(int)(NW / 1024), 256, 0, stream>>>(Wo, wob, (int)NW);

    // 2) QKV projections: (4096x2048) = xb @ W^T
    const int M = BATCH * SEQ;  // 4096
    dim3 gg(DMODEL / BN, M / BM);   // (16, 16)
    gemm_nt_bf16<<<gg, 256, 0, stream>>>(xb, wqb, Qf, M, DMODEL, DMODEL);
    gemm_nt_bf16<<<gg, 256, 0, stream>>>(xb, wkb, Kf, M, DMODEL, DMODEL);
    gemm_nt_bf16<<<gg, 256, 0, stream>>>(xb, wvb, Vf, M, DMODEL, DMODEL);

    // 3) RoPE + RMS-norm + bf16 cast
    rope_norm_cast<<<(BATCH * SEQ * NH) / 8, 256, 0, stream>>>(
        Qf, Kf, Vf, cosT, sinT, Qb, Kb, Vb);

    // 4) flash attention -> Yb (bf16)
    dim3 ga(SEQ / 128, NH, BATCH);  // (16, 16, 2)
    flash_attn<<<ga, 256, 0, stream>>>(Qb, Kb, Vb, Yb);

    // 5) output projection -> d_out (fp32)
    gemm_nt_bf16<<<gg, 256, 0, stream>>>(Yb, wob, out, M, DMODEL, DMODEL);
}
